// MultiHeadAttention_47347719471561
// MI455X (gfx1250) — compile-verified, matrix operand order checked by reading, not run on hardware
//
#include <hip/hip_runtime.h>

// ---- problem constants ----
#define SEQL   2048
#define BATCHN 2
#define NHEADS 16
#define DKH    64
#define DMODEL 1024
#define MROWS  (SEQL * BATCHN) // 4096

typedef __bf16 bf16;
typedef bf16  bf16x8 __attribute__((ext_vector_type(8)));
typedef bf16  v16bf  __attribute__((ext_vector_type(16)));
typedef float v8f    __attribute__((ext_vector_type(8)));
typedef float f32x8  __attribute__((ext_vector_type(8)));

union FragU { v16bf v; bf16x8 h[2]; };

// fp32 -> bf16 RNE via integer ALU (avoids relying on f32->bf16 fptrunc codegen)
__device__ inline bf16 f2bf(float f) {
  unsigned u = __float_as_uint(f);
  u += 0x7fffu + ((u >> 16) & 1u);
  unsigned short hs = (unsigned short)(u >> 16);
  bf16 r;
  __builtin_memcpy(&r, &hs, sizeof(r));
  return r;
}

__device__ inline v8f vzero8() {
  v8f z;
  for (int i = 0; i < 8; ++i) z[i] = 0.f;
  return z;
}

// A/B fragment (16x32 bf16, MxK row-major with leading dim `ld`):
//   lanes 0-15  hold row (row0+lr), K = {k0..k0+7} U {k0+16..k0+23}
//   lanes 16-31 hold row (row0+lr), K = {k0+8..k0+15} U {k0+24..k0+31}
// -> two contiguous 16B chunks per lane.
__device__ inline v16bf load_frag_bf16(const bf16* p, int ld, int row0, int k0, int lane) {
  const int half = (lane >> 4) & 1;
  const int lr   = lane & 15;
  const bf16* base = p + (size_t)(row0 + lr) * ld + k0 + half * 8;
  FragU u;
  u.h[0] = *(const bf16x8*)(base);
  u.h[1] = *(const bf16x8*)(base + 16);
  return u.v;
}

__device__ inline v16bf load_frag_f32(const float* p, int ld, int row0, int k0, int lane) {
  const int half = (lane >> 4) & 1;
  const int lr   = lane & 15;
  const float* base = p + (size_t)(row0 + lr) * ld + k0 + half * 8;
  f32x8 c0 = *(const f32x8*)(base);
  f32x8 c1 = *(const f32x8*)(base + 16);
  FragU u;
  for (int i = 0; i < 8; ++i) { u.h[0][i] = f2bf(c0[i]); u.h[1][i] = f2bf(c1[i]); }
  return u.v;
}

__device__ inline v8f wmma_bf16(v16bf a, v16bf b, v8f c) {
  // D(16x16 f32) = A(16x32 bf16) * B(32x16 bf16) + C
  return __builtin_amdgcn_wmma_f32_16x16x32_bf16(false, a, false, b, (short)0, c, false, false);
}

// ---------------------------------------------------------------------------
// GEMM: C[m,n] = sum_k A[m,k] * W[n,k] + bias[n]   (torch Linear: x @ W^T + b)
// M=4096, N=1024, K=1024. Wave computes a 16x64 tile (A fragment reused 4x).
// MODE 0: A fp32, out = bf16 [B,H,S,DK]      (Q, K projections)
// MODE 1: A fp32, out = bf16 [B,H,DK,S]      (V projection, transposed)
// MODE 2: A bf16, out = fp32 [M,N]           (output projection -> d_out)
// ---------------------------------------------------------------------------
template <int MODE>
__global__ __launch_bounds__(256) void gemm16x64_wmma(const void* __restrict__ Aray,
                                                      const float* __restrict__ W,
                                                      const float* __restrict__ bias,
                                                      void* __restrict__ Out) {
  const int lane  = threadIdx.x & 31;
  const int wave  = threadIdx.x >> 5;
  const int wtile = blockIdx.x * 8 + wave;   // 4096 wave tiles
  const int m0 = (wtile >> 4) * 16;          // 256 M tiles
  const int n0 = (wtile & 15) * 64;          // 16 N supertiles of 64

  v8f acc[4];
  for (int t = 0; t < 4; ++t) acc[t] = vzero8();

  for (int k0 = 0; k0 < DMODEL; k0 += 32) {
    v16bf a;
    if (MODE == 2) a = load_frag_bf16((const bf16*)Aray, DMODEL, m0, k0, lane);
    else           a = load_frag_f32((const float*)Aray, DMODEL, m0, k0, lane);
#pragma unroll
    for (int nt = 0; nt < 4; ++nt) {
      v16bf b = load_frag_f32(W, DMODEL, n0 + nt * 16, k0, lane);
      acc[nt] = wmma_bf16(a, b, acc[nt]);
    }
  }

  // C layout: VGPR j, lanes 0-15 -> M=j, lanes 16-31 -> M=j+8; N = lane&15
  const int half = (lane >> 4) & 1;
  const int lr   = lane & 15;
#pragma unroll
  for (int nt = 0; nt < 4; ++nt) {
    const int col = n0 + nt * 16 + lr;
    const float bv = bias[col];
#pragma unroll
    for (int j = 0; j < 8; ++j) {
      const int mrow = m0 + j + 8 * half;
      const float v = acc[nt][j] + bv;
      if (MODE == 0) {
        const int s = mrow >> 1, bb = mrow & 1;       // m = s*B + b
        const int h = col >> 6,  d  = col & 63;
        ((bf16*)Out)[(((size_t)(bb * NHEADS + h)) * SEQL + s) * DKH + d] = f2bf(v);
      } else if (MODE == 1) {
        const int s = mrow >> 1, bb = mrow & 1;
        const int h = col >> 6,  d  = col & 63;
        ((bf16*)Out)[(((size_t)(bb * NHEADS + h)) * DKH + d) * SEQL + s] = f2bf(v);
      } else {
        ((float*)Out)[(size_t)mrow * DMODEL + col] = v;  // == [S,B,D] flat
      }
    }
  }
}

// ---------------------------------------------------------------------------
// Flash attention: per (b,h), per 16-row query tile per wave.
// Never materializes the 32x2048x2048 score tensor (saves ~1 GB of HBM traffic;
// pos_bias's 256 MB stream is the mandatory-traffic floor).
// ---------------------------------------------------------------------------
__global__ __launch_bounds__(256) void attn_flash_wmma(const bf16* __restrict__ Q,
                                                       const bf16* __restrict__ K,
                                                       const bf16* __restrict__ Vt,
                                                       const float* __restrict__ pos_bias,
                                                       const int* __restrict__ mask,
                                                       bf16* __restrict__ ctx) {
  __shared__ __align__(16) bf16 Plds[8][16][32];  // per-wave 16x32 P staging (8 KB)

  const int lane = threadIdx.x & 31;
  const int wave = threadIdx.x >> 5;
  const int half = (lane >> 4) & 1;
  const int lr   = lane & 15;

  const int qBlk = blockIdx.x & 15;   // 16 blocks of 128 queries
  const int bh   = blockIdx.x >> 4;   // 0..31
  const int bb   = bh >> 4;
  const int h    = bh & 15;

  const bf16*  Qh  = Q  + (size_t)bh * SEQL * DKH;
  const bf16*  Kh  = K  + (size_t)bh * SEQL * DKH;
  const bf16*  Vth = Vt + (size_t)bh * DKH * SEQL;
  const float* pb  = pos_bias + (size_t)h * SEQL * SEQL;

  const int q0 = qBlk * 128 + wave * 16;

  const v16bf qf0 = load_frag_bf16(Qh, DKH, q0, 0,  lane);  // d = 0..31
  const v16bf qf1 = load_frag_bf16(Qh, DKH, q0, 32, lane);  // d = 32..63

  float mrow[8], lrow[8];
  for (int j = 0; j < 8; ++j) { mrow[j] = -3.0e38f; lrow[j] = 0.f; }
  v8f Oacc[4];
  for (int t = 0; t < 4; ++t) Oacc[t] = vzero8();

  for (int kc = 0; kc < SEQL; kc += 32) {
    // ---- scores: 16 queries x 32 keys, K-dim = 64 (two WMMAs per 16-key tile)
    v8f s0 = vzero8(), s1 = vzero8();
    {
      v16bf kf0 = load_frag_bf16(Kh, DKH, kc, 0,  lane);
      v16bf kf1 = load_frag_bf16(Kh, DKH, kc, 32, lane);
      s0 = wmma_bf16(qf0, kf0, s0);
      s0 = wmma_bf16(qf1, kf1, s0);
    }
    {
      v16bf kf0 = load_frag_bf16(Kh, DKH, kc + 16, 0,  lane);
      v16bf kf1 = load_frag_bf16(Kh, DKH, kc + 16, 32, lane);
      s1 = wmma_bf16(qf0, kf0, s1);
      s1 = wmma_bf16(qf1, kf1, s1);
    }
    if (kc + 32 < SEQL)
      __builtin_prefetch(Kh + (size_t)(kc + 32) * DKH, 0, 0);  // global_prefetch_b8

    // ---- scale + bias + mask + online softmax (row = j + 8*half across 16 lanes)
#pragma unroll
    for (int j = 0; j < 8; ++j) {
      const int r = q0 + j + 8 * half;          // global query index s
      const size_t rowoff = (size_t)r * SEQL;
      const int c0 = kc + lr;
      const int c1 = kc + 16 + lr;
      float sc0 = s0[j] * 0.125f + pb[rowoff + c0];   // 1/sqrt(64) = 0.125
      float sc1 = s1[j] * 0.125f + pb[rowoff + c1];
      if (mask[rowoff + c0] == 0) sc0 = -1.0e9f;
      if (mask[rowoff + c1] == 0) sc1 = -1.0e9f;

      float mx = fmaxf(sc0, sc1);
      for (int off = 1; off < 16; off <<= 1) mx = fmaxf(mx, __shfl_xor(mx, off, 32));
      const float mnew  = fmaxf(mrow[j], mx);
      const float alpha = __expf(mrow[j] - mnew);
      const float p0 = __expf(sc0 - mnew);
      const float p1 = __expf(sc1 - mnew);
      float ps = p0 + p1;
      for (int off = 1; off < 16; off <<= 1) ps += __shfl_xor(ps, off, 32);
      lrow[j] = lrow[j] * alpha + ps;
      mrow[j] = mnew;
#pragma unroll
      for (int t = 0; t < 4; ++t) Oacc[t][j] *= alpha;

      Plds[wave][j + 8 * half][lr]      = f2bf(p0);
      Plds[wave][j + 8 * half][16 + lr] = f2bf(p1);
    }
    __syncthreads();  // C-layout writes -> A-layout reads (cross-lane in LDS)

    FragU pf;
    const bf16* prow = &Plds[wave][lr][0];
    pf.h[0] = *(const bf16x8*)(prow + half * 8);
    pf.h[1] = *(const bf16x8*)(prow + 16 + half * 8);

    // ---- O += P(16x32) @ V(32x64); Vt rows are d, so B fragments are row loads
#pragma unroll
    for (int t = 0; t < 4; ++t) {
      v16bf vf = load_frag_bf16(Vth, SEQL, t * 16, kc, lane);
      Oacc[t] = wmma_bf16(pf.v, vf, Oacc[t]);
    }
    __syncthreads();  // protect P staging before next iteration's writes
  }

  // ---- normalize and emit context as bf16 rows m = s*B + b, col = h*64 + d
#pragma unroll
  for (int t = 0; t < 4; ++t) {
#pragma unroll
    for (int j = 0; j < 8; ++j) {
      const int s   = q0 + j + 8 * half;
      const int m   = s * BATCHN + bb;
      const int col = h * DKH + t * 16 + lr;
      ctx[(size_t)m * DMODEL + col] = f2bf(Oacc[t][j] / lrow[j]);
    }
  }
}

// ---------------------------------------------------------------------------
extern "C" void kernel_launch(void* const* d_in, const int* in_sizes, int n_in,
                              void* d_out, int out_size, void* d_ws, size_t ws_size,
                              hipStream_t stream) {
  (void)in_sizes; (void)n_in; (void)out_size; (void)ws_size;

  const float* query = (const float*)d_in[0];
  const float* key   = (const float*)d_in[1];
  const float* value = (const float*)d_in[2];
  const int*   mask  = (const int*)d_in[3];
  const float* pbias = (const float*)d_in[4];
  const float* Wq = (const float*)d_in[5];
  const float* bq = (const float*)d_in[6];
  const float* Wk = (const float*)d_in[7];
  const float* bk = (const float*)d_in[8];
  const float* Wv = (const float*)d_in[9];
  const float* bv = (const float*)d_in[10];
  const float* Wo = (const float*)d_in[11];
  const float* bo = (const float*)d_in[12];

  char* ws = (char*)d_ws;
  const size_t te = (size_t)BATCHN * NHEADS * SEQL * DKH;  // 4M elems, 8 MB bf16
  bf16* Qb  = (bf16*)(ws);
  bf16* Kb  = (bf16*)(ws + te * 2);
  bf16* Vtb = (bf16*)(ws + te * 4);
  bf16* ctx = (bf16*)(ws + te * 6);   // total 32 MB workspace

  dim3 grid(512), blk(256);
  gemm16x64_wmma<0><<<grid, blk, 0, stream>>>(query, Wq, bq, Qb);
  gemm16x64_wmma<0><<<grid, blk, 0, stream>>>(key,   Wk, bk, Kb);
  gemm16x64_wmma<1><<<grid, blk, 0, stream>>>(value, Wv, bv, Vtb);
  attn_flash_wmma<<<grid, blk, 0, stream>>>(Qb, Kb, Vtb, pbias, mask, ctx);
  gemm16x64_wmma<2><<<grid, blk, 0, stream>>>(ctx, Wo, bo, d_out);
}
